// Attention_65300682768972
// MI455X (gfx1250) — matrix operand order repaired and checked by reading, not compiled, not run on hardware
//
#include <hip/hip_runtime.h>

#define BB 64
#define SS 4096
#define HH 256

typedef __attribute__((ext_vector_type(16))) __bf16 v16bf;
typedef __attribute__((ext_vector_type(8)))  float  v8f;
typedef __attribute__((ext_vector_type(4)))  float  v4f;

// ---- dynamic LDS layout (bytes) ----
#define WK_OFF   0               // __bf16[65536]  = 131072 B (Wk, row-major)
#define Q_OFF    131072          // float[256]     (q = query@Wq^T)
#define V_OFF    132096          // float[256]     (v vector)
#define E_OFF    133120          // float[4096]    (energies -> weights)
#define RED_OFF  149504          // float[16]      (block reductions)
#define QIN_OFF  149568          // float[256]     (staged query row)
#define SMEM_BYTES 150592

// Branch-free tanh: 1 - 2/(exp(2x)+1).  v_exp + v_rcp, saturates to +/-1.
__device__ __forceinline__ float fast_tanh(float x) {
    float ex = __expf(2.f * x);
    return 1.f - 2.f * __builtin_amdgcn_rcpf(ex + 1.f);
}

// Build bf16 A-fragment for V_WMMA_F32_16X16X32_BF16 from fp32 row data.
// Per ISA layout (16-bit A 16x32): lanes 0-15 take K = 0..7 (V0..V3) and
// K = 16..23 (V4..V7); lanes 16-31 take K = 8..15 and 24..31. Caller passes
// p = row + kstep*32 + khalf*8, so we read [p..p+7] and [p+16..p+23].
__device__ __forceinline__ v16bf load_a_frag(const float* __restrict__ p) {
    v4f f0 = *(const v4f*)(p + 0);
    v4f f1 = *(const v4f*)(p + 4);
    v4f f2 = *(const v4f*)(p + 16);
    v4f f3 = *(const v4f*)(p + 20);
    v16bf r;
    r[0]  = (__bf16)f0.x; r[1]  = (__bf16)f0.y; r[2]  = (__bf16)f0.z; r[3]  = (__bf16)f0.w;
    r[4]  = (__bf16)f1.x; r[5]  = (__bf16)f1.y; r[6]  = (__bf16)f1.z; r[7]  = (__bf16)f1.w;
    r[8]  = (__bf16)f2.x; r[9]  = (__bf16)f2.y; r[10] = (__bf16)f2.z; r[11] = (__bf16)f2.w;
    r[12] = (__bf16)f3.x; r[13] = (__bf16)f3.y; r[14] = (__bf16)f3.z; r[15] = (__bf16)f3.w;
    return r;
}

__launch_bounds__(256, 1)
__global__ void attn_additive_fused(const float* __restrict__ query,
                                    const float* __restrict__ keys,
                                    const float* __restrict__ Wq,
                                    const float* __restrict__ Wk,
                                    const float* __restrict__ vvec,
                                    float* __restrict__ out_ctx,
                                    float* __restrict__ out_w)
{
    extern __shared__ char smem[];
    __bf16* wk_lds = (__bf16*)(smem + WK_OFF);
    float*  q_lds  = (float*)(smem + Q_OFF);
    float*  v_lds  = (float*)(smem + V_OFF);
    float*  e_lds  = (float*)(smem + E_OFF);
    float*  red    = (float*)(smem + RED_OFF);
    float*  qin    = (float*)(smem + QIN_OFF);

    const int t     = threadIdx.x;          // 0..255
    const int b     = blockIdx.x;           // batch
    const int lane  = t & 31;
    const int wid   = t >> 5;               // 0..7
    const int khalf = lane >> 4;            // 0 or 1 (lane half)
    const int nsub  = lane & 15;            // column / row-in-tile index

    // ---- stage query row and v ----
    qin[t]   = query[(size_t)b * HH + t];
    v_lds[t] = vvec[t];
    __syncthreads();

    // ---- q[b] = query[b] @ Wq^T  (thread t computes q[t]) ----
    {
        float acc = 0.f;
        const float* wr = Wq + (size_t)t * HH;
        #pragma unroll 8
        for (int k = 0; k < HH; ++k) acc += qin[k] * wr[k];
        q_lds[t] = acc;
    }

    // ---- stage Wk (row-major HxH) as bf16 into LDS ----
    for (int i = t; i < HH * HH; i += 256)
        wk_lds[i] = (__bf16)Wk[i];
    __syncthreads();

    // ---- main loop: k = keys @ Wk^T via WMMA, fused energy = tanh(q+k)@v ----
    // Each wave owns 512 rows of S in 32 tiles of 16 rows.
    const int rows_per_wave = SS / 8;       // 512
    for (int tile = 0; tile < rows_per_wave / 16; ++tile) {
        const int srow = wid * rows_per_wave + tile * 16;
        // A layout: this lane supplies row m = nsub of the tile.
        const float* arow = keys + ((size_t)b * SS + srow + nsub) * HH;

        // Preload all 8 K-step A fragments (K = 256 = 8 * 32)
        v16bf af[8];
        #pragma unroll
        for (int ks = 0; ks < 8; ++ks)
            af[ks] = load_a_frag(arow + ks * 32 + khalf * 8);

        // Prefetch next tile's keys rows while this tile's WMMAs run.
        if (tile + 1 < rows_per_wave / 16)
            __builtin_prefetch(arow + 16 * HH, 0, 0);

        float eacc[8] = {0.f,0.f,0.f,0.f,0.f,0.f,0.f,0.f};

        for (int nt = 0; nt < HH / 16; ++nt) {
            const int ng = nt * 16 + nsub;  // global output column n
            v8f c = {};
            #pragma unroll
            for (int ks = 0; ks < 8; ++ks) {
                // B layout (32x16 bf16): lane holds column n = nsub,
                // K = khalf*8 + 0..7 and 16 + khalf*8 + 0..7 of
                // Wk row n (B[k][n] = Wk[n][k]).
                typedef __attribute__((ext_vector_type(8))) __bf16 v8bf_;
                const __bf16* bp = wk_lds + (size_t)ng * HH + ks * 32 + khalf * 8;
                v8bf_ b0 = *(const v8bf_*)(bp);
                v8bf_ b1 = *(const v8bf_*)(bp + 16);
                v16bf bf = __builtin_shufflevector(b0, b1, 0,1,2,3,4,5,6,7,8,9,10,11,12,13,14,15);
                c = __builtin_amdgcn_wmma_f32_16x16x32_bf16(
                        false, af[ks], false, bf, (short)0, c, false, false);
            }
            // C layout: c[j] is row m = j + 8*khalf, col n = nsub.
            const float qn = q_lds[ng];
            const float vn = v_lds[ng];
            #pragma unroll
            for (int j = 0; j < 8; ++j)
                eacc[j] += fast_tanh(qn + c[j]) * vn;
        }

        // Reduce over the 16 columns held by each 16-lane group.
        #pragma unroll
        for (int j = 0; j < 8; ++j) {
            #pragma unroll
            for (int off = 8; off >= 1; off >>= 1)
                eacc[j] += __shfl_xor(eacc[j], off, 16);
        }
        if (nsub == 0) {
            #pragma unroll
            for (int j = 0; j < 8; ++j)
                e_lds[srow + khalf * 8 + j] = eacc[j];
        }
    }
    __syncthreads();

    // ---- softmax over S = 4096 (block-wide) ----
    float lmax = -3.4e38f;
    for (int i = t; i < SS; i += 256) lmax = fmaxf(lmax, e_lds[i]);
    #pragma unroll
    for (int off = 16; off >= 1; off >>= 1)
        lmax = fmaxf(lmax, __shfl_xor(lmax, off, 32));
    if (lane == 0) red[wid] = lmax;
    __syncthreads();
    float gmax = red[0];
    #pragma unroll
    for (int i = 1; i < 8; ++i) gmax = fmaxf(gmax, red[i]);

    float lsum = 0.f;
    for (int i = t; i < SS; i += 256) {
        float ex = __expf(e_lds[i] - gmax);
        e_lds[i] = ex;
        lsum += ex;
    }
    #pragma unroll
    for (int off = 16; off >= 1; off >>= 1)
        lsum += __shfl_xor(lsum, off, 32);
    if (lane == 0) red[8 + wid] = lsum;
    __syncthreads();
    float gsum = 0.f;
    #pragma unroll
    for (int i = 0; i < 8; ++i) gsum += red[8 + i];
    const float inv = 1.f / gsum;

    for (int i = t; i < SS; i += 256) {
        float w = e_lds[i] * inv;
        e_lds[i] = w;
        out_w[(size_t)b * SS + i] = w;
    }
    __syncthreads();

    // ---- context[b, h] = sum_s w[s] * keys[b, s, h]  (h = t, coalesced) ----
    {
        float acc = 0.f;
        const float* kb = keys + (size_t)b * SS * HH + t;
        #pragma unroll 8
        for (int s = 0; s < SS; ++s)
            acc += e_lds[s] * kb[(size_t)s * HH];
        out_ctx[(size_t)b * HH + t] = acc;
    }
}

extern "C" void kernel_launch(void* const* d_in, const int* in_sizes, int n_in,
                              void* d_out, int out_size, void* d_ws, size_t ws_size,
                              hipStream_t stream) {
    (void)in_sizes; (void)n_in; (void)out_size; (void)d_ws; (void)ws_size;
    const float* query = (const float*)d_in[0];   // (B,1,H)
    const float* keys  = (const float*)d_in[1];   // (B,S,H)
    const float* Wq    = (const float*)d_in[2];   // (H,H)
    const float* Wk    = (const float*)d_in[3];   // (H,H)
    const float* vvec  = (const float*)d_in[4];   // (H,)

    float* out_ctx = (float*)d_out;               // (B,1,H) first
    float* out_w   = out_ctx + BB * HH;           // then (B,1,S)

    attn_additive_fused<<<BB, 256, SMEM_BYTES, stream>>>(
        query, keys, Wq, Wk, vvec, out_ctx, out_w);
}
